// MRA_1812476199110
// MI455X (gfx1250) — matrix operand order repaired and checked by reading, not compiled, not run on hardware
//
#include <hip/hip_runtime.h>
#include <hip/hip_bf16.h>

// ---------------------------------------------------------------------------
// MRA fused kernel for MI455X (gfx1250).
// One workgroup per (batch, channel) plane:
//   TDM async load 192x192 plane -> LDS (147KB), v_wmma for axis-aligned convs,
//   VALU gather for diagonal convs, BN+sigmoid, fused upsample*multiply store.
// HBM traffic = read x once + write out once = 302 MB ~= 13.0us @ 23.3 TB/s.
// All conv-stage LDS reads are unconditional (zero-padded halo + zero-padded
// band-weight tables) so loads batch under a single s_wait_dscnt.
// ---------------------------------------------------------------------------

#define BB   16
#define CC   64
#define HH   192
#define WW   192
#define HT   64            // pooled spatial size
#define XTPD 82            // padded x_tem row stride (81 rows x 82)
#define XTPN (81 * XTPD)   // 6642 floats
#define EPSI 1e-5f

typedef __attribute__((ext_vector_type(16))) _Float16     v16h;
typedef __attribute__((ext_vector_type(8)))  float        v8f;
typedef __attribute__((ext_vector_type(4)))  unsigned int v4u;
typedef __attribute__((ext_vector_type(8)))  int          v8i;
typedef __attribute__((ext_vector_type(4)))  int          v4i;

// LDS layout (floats)
#define XS_OFF    0
#define XTP_OFF   (XS_OFF + HH * WW)          // 36864
#define GATE_OFF  (XTP_OFF + XTPN)            // + 6642
#define M_OFF     (GATE_OFF + HT * HT)        // + 4096
#define HS_OFF    (M_OFF + 25 * WW)           // + 4800
#define WB1_OFF   (HS_OFF + 25 * HT)          // + 1600   wb1[48][3]
#define WB2_OFF   (WB1_OFF + 144)             //          wb2[3][48]
#define WH2_OFF   (WB2_OFF + 144)
#define WV2_OFF   (WH2_OFF + 33)
#define BN_OFF    (WV2_OFF + 33)
#define SMEM_FLOATS (BN_OFF + 2)              // 54358
#define SMEM_BYTES  (SMEM_FLOATS * 4)         // ~217.4 KB < 320 KB WGP LDS

// padded x_tem accessor: valid for y,x in [-6, 74]
#define XTP(y, x) xtp[((y) + 6) * XTPD + ((x) + 6)]

__global__ __launch_bounds__(256, 1)
void mra_fused_gfx1250(const float* __restrict__ x,
                       const float* __restrict__ wh1g, const float* __restrict__ wv1g,
                       const float* __restrict__ wh2g, const float* __restrict__ wv2g,
                       const float* __restrict__ bng,  const float* __restrict__ bnb,
                       const float* __restrict__ bnm,  const float* __restrict__ bnv,
                       float* __restrict__ out)
{
  extern __shared__ __align__(16) float smem[];
  float* xs   = smem + XS_OFF;     // 192x192 input plane
  float* xtp  = smem + XTP_OFF;    // padded 81x81 pooled tile (stride 82)
  float* gate = smem + GATE_OFF;   // 64x64 gate
  float* Ms   = smem + M_OFF;      // 25x192 maxpool strip
  float* Hs   = smem + HS_OFF;     // 25x64 horiz-blur strip
  float* wb1  = smem + WB1_OFF;    // padded band: wb1[(d+16)*3 + bb], d in [-16,31]
  float* wb2  = smem + WB2_OFF;    // padded band: wb2[bb*48 + d + 16]
  float* wh2  = smem + WH2_OFF;
  float* wv2  = smem + WV2_OFF;
  float* bns  = smem + BN_OFF;

  const int tid  = threadIdx.x;
  const int chan = blockIdx.x;           // b*C + c
  const int c    = chan & (CC - 1);
  const size_t base = (size_t)chan * (HH * WW);

  // ---- Stage 0: bulk plane load HBM -> LDS via Tensor Data Mover -----------
#if defined(__gfx1250__) && __has_builtin(__builtin_amdgcn_tensor_load_to_lds)
  if (tid < 32) {                        // single wave issues the TDM op
    const unsigned long long ga  = (unsigned long long)(x + base);
    const unsigned int       glo = (unsigned int)ga;
    const unsigned int       ghi = (unsigned int)(ga >> 32);
    const unsigned int       lds = (unsigned int)(unsigned long long)xs; // LDS byte offset
    // D# group0: count=1 | lds_addr | global_addr[56:0] | type=2
    v4u g0 = { 1u, lds, glo, (ghi & 0x01FFFFFFu) | (2u << 30) };
    // D# group1: data_size=4B; tensor 192x192, tile 192x192, stride0=192
    v8i g1 = { (int)(2u << 16),          // wg_mask=0, data_size=2 (4B)
               (int)(WW << 16),          // tensor_dim0[15:0] -> bits63:48
               (int)(HH << 16),          // tensor_dim0 hi=0 | tensor_dim1 lo
               (int)(WW << 16),          // tensor_dim1 hi=0 | tile_dim0=192
               (int)HH,                  // tile_dim1=192 | tile_dim2=0
               (int)WW,                  // tensor_dim0_stride = 192
               0, 0 };
    v4i g2 = {0, 0, 0, 0};
    v4i g3 = {0, 0, 0, 0};
    v8i g4 = {0, 0, 0, 0, 0, 0, 0, 0};
    __builtin_amdgcn_tensor_load_to_lds(g0, g1, g2, g3, g4, 0);
#if __has_builtin(__builtin_amdgcn_s_wait_tensorcnt)
    __builtin_amdgcn_s_wait_tensorcnt(0);
#endif
  }
#else
  for (int i4 = tid; i4 < (HH * WW) / 4; i4 += 256)
    reinterpret_cast<float4*>(xs)[i4] =
        reinterpret_cast<const float4*>(x + base)[i4];
#endif

  // zero the padded pooled tile and padded weight tables (wb1||wb2 contiguous)
  for (int i = tid; i < XTPN; i += 256) xtp[i] = 0.f;
  for (int i = tid; i < 288; i += 256) wb1[i] = 0.f;
  __syncthreads();

  // per-channel weights (centers of padded tables) + BN constants
  if (tid < 33) {                                    // wh1: (11,3) row-major
    const int d = tid / 3, bb = tid - 3 * d;
    wb1[(d + 16) * 3 + bb] = wh1g[c * 33 + tid];
  } else if (tid < 66) {                             // wv1: (3,11) row-major
    const int t = tid - 33;
    const int bb = t / 11, d = t - 11 * bb;
    wb2[bb * 48 + d + 16] = wv1g[c * 33 + t];
  } else if (tid < 99) {
    wh2[tid - 66] = wh2g[c * 33 + (tid - 66)];
  } else if (tid < 132) {
    wv2[tid - 99] = wv2g[c * 33 + (tid - 99)];
  }
  if (tid == 255) {
    const float inv = bng[c] * rsqrtf(bnv[c] + EPSI);
    bns[0] = inv;
    bns[1] = bnb[c] - bnm[c] * inv;
  }
  __syncthreads();

  // ---- Stage 1: maxpool3x3(s1,p1) + blurpool(4tap,s3,reflect(1,2)) --------
  // blur taps (per axis): rows/cols {3k-1, 3k, 3k+1, 3k+2}, weights {1,3,3,1}/8,
  // reflect maps index -1 -> 1 (the +2 pad side is never sampled).
  for (int s = 0; s < 8; ++s) {
    const int y0 = s * 8;
    const int t0 = 3 * y0 - 1;                       // first maxpool row in strip
    for (int idx = tid; idx < 25 * WW; idx += 256) { // Ms = 3x3 max of xs
      const int tt = idx / WW, ccx = idx - tt * WW;
      const int t = t0 + tt;
      if (t >= 0 && t < HH) {
        const int r0 = t > 0 ? t - 1 : 0,      r1 = t < HH - 1 ? t + 1 : HH - 1;
        const int c0 = ccx > 0 ? ccx - 1 : 0,  c1 = ccx < WW - 1 ? ccx + 1 : WW - 1;
        float m = -1e30f;
        for (int r = r0; r <= r1; ++r)
          for (int q = c0; q <= c1; ++q)
            m = fmaxf(m, xs[r * WW + q]);
        Ms[idx] = m;
      }
    }
    __syncthreads();
    for (int idx = tid; idx < 25 * HT; idx += 256) { // Hs = horiz 4-tap, stride 3
      const int tt = idx / HT, xc = idx - tt * HT;
      const int t = t0 + tt;
      if (t >= 0 && t < HH) {
        int q = 3 * xc - 1; if (q < 0) q = 1;        // reflect
        const float a0 = Ms[tt * WW + q];
        const float a1 = Ms[tt * WW + 3 * xc + 0];
        const float a2 = Ms[tt * WW + 3 * xc + 1];
        const float a3 = Ms[tt * WW + 3 * xc + 2];
        Hs[idx] = (a0 + 3.f * a1 + 3.f * a2 + a3) * 0.125f;
      }
    }
    __syncthreads();
    for (int idx = tid; idx < 8 * HT; idx += 256) {  // vert 4-tap -> x_tem rows
      const int yy = idx / HT, xc = idx - yy * HT;
      const int y = y0 + yy;
      float acc = 0.f;
      const float fw[4] = {1.f, 3.f, 3.f, 1.f};
      #pragma unroll
      for (int i = 0; i < 4; ++i) {
        int p = 3 * y + i - 1; if (p < 0) p = 1;     // reflect
        acc += fw[i] * Hs[(p - t0) * HT + xc];
      }
      XTP(y, xc) = acc * 0.125f;
    }
    __syncthreads();
  }

  // ---- Stage 2: 4-branch depthwise conv + BN + sigmoid --------------------
  const float bninv = bns[0], bnsh = bns[1];
  const int lane = tid & 31;
  const int wv   = tid >> 5;          // 8 waves
  const int g    = lane >> 4;         // lane group (0/1)
  const int ln   = lane & 15;

  // block-invariant weight fragments (band matrices), built once per wave
  v16h a1f[3], b2f[3];
  #pragma unroll
  for (int bb = 0; bb < 3; ++bb) {
    #pragma unroll
    for (int e = 0; e < 16; ++e) {
      const int KA = e + 8 * g + ((e >= 8) ? 8 : 0);
      const int KB = e + 16 * g;
      a1f[bb][e] = (_Float16)wb1[(KA - ln + 16) * 3 + bb];   // wh1 band (A frag)
      b2f[bb][e] = (_Float16)wb2[bb * 48 + (KB - ln + 16)];  // wv1 band (B frag)
    }
  }

  for (int bi = 2 * wv; bi < 2 * wv + 2; ++bi) {     // 16 blocks, 2 per wave
    const int R  = (bi >> 2) * 16;
    const int Cb = (bi & 3) * 16;
    v8f acc = {0.f, 0.f, 0.f, 0.f, 0.f, 0.f, 0.f, 0.f};

    #pragma unroll
    for (int bb = 0; bb < 3; ++bb) {                 // b = bb - 1
      const int b = bb - 1;
      v16h b1, a2;
      #pragma unroll
      for (int e = 0; e < 16; ++e) {
        const int KA = e + 8 * g + ((e >= 8) ? 8 : 0);
        const int KB = e + 16 * g;
        // branch1: out[i,j] = sum_k wh1[k-i][b] * xt(R+k-5, Cb+j+b)
        b1[e] = (_Float16)XTP(R + KA - 5, Cb + ln + b);
        // branch2: out[i,j] = sum_k xt(R+i+b, Cb+k-5) * wv1[b][k-j]
        a2[e] = (_Float16)XTP(R + ln + b, Cb + KB - 5);
      }
      acc = __builtin_amdgcn_wmma_f32_16x16x32_f16(false, a1f[bb], false, b1,
                                                   (short)0, acc, false, false);
      acc = __builtin_amdgcn_wmma_f32_16x16x32_f16(false, a2, false, b2f[bb],
                                                   (short)0, acc, false, false);
    }

    // diagonal branches (h/v "transform" paths) + BN + sigmoid
    #pragma unroll
    for (int r = 0; r < 8; ++r) {
      const int y  = R + r + 8 * g;                  // C/D layout: M = r + 8*g
      const int xq = Cb + ln;
      float sacc = acc[r];
      for (int a = -5; a <= 5; ++a) {
        #pragma unroll
        for (int bb = 0; bb < 3; ++bb) {
          const int b = bb - 1;
          sacc += wh2[(a + 5) * 3 + bb] * XTP(y + a,     xq + b - a);
          sacc += wv2[bb * 11 + (a + 5)] * XTP(y + b - a, xq + a);
        }
      }
      const float t = sacc * bninv + bnsh;
      gate[y * HT + xq] = 1.f / (1.f + __expf(-t));
    }
  }
  __syncthreads();

  // ---- Stage 3: out = xs * nearest-upsample(gate), float4 streaming -------
  float* outp = out + base;
  for (int i4 = tid; i4 < (HH * WW) / 4; i4 += 256) {
    const int li = i4 * 4;
    const int Y  = li / WW;
    const int X  = li - Y * WW;
    const float* xr = xs + Y * WW;
    const float* gr = gate + (Y / 3) * HT;
    float4 v;
    v.x = xr[X + 0] * gr[(X + 0) / 3];
    v.y = xr[X + 1] * gr[(X + 1) / 3];
    v.z = xr[X + 2] * gr[(X + 2) / 3];
    v.w = xr[X + 3] * gr[(X + 3) / 3];
    reinterpret_cast<float4*>(outp)[i4] = v;
  }
}

extern "C" void kernel_launch(void* const* d_in, const int* in_sizes, int n_in,
                              void* d_out, int out_size, void* d_ws, size_t ws_size,
                              hipStream_t stream) {
  (void)in_sizes; (void)n_in; (void)out_size; (void)d_ws; (void)ws_size;
  const float* x    = (const float*)d_in[0];
  const float* wh1  = (const float*)d_in[1];
  const float* wv1  = (const float*)d_in[2];
  const float* wh2  = (const float*)d_in[3];
  const float* wv2  = (const float*)d_in[4];
  const float* bngm = (const float*)d_in[5];
  const float* bnbt = (const float*)d_in[6];
  const float* bnmn = (const float*)d_in[7];
  const float* bnvr = (const float*)d_in[8];
  float* out = (float*)d_out;

  // Opt in to >64KB dynamic LDS (gfx1250 WGP supports 320KB per workgroup).
  (void)hipFuncSetAttribute((const void*)mra_fused_gfx1250,
                            hipFuncAttributeMaxDynamicSharedMemorySize,
                            (int)SMEM_BYTES);

  mra_fused_gfx1250<<<BB * CC, 256, SMEM_BYTES, stream>>>(
      x, wh1, wv1, wh2, wv2, bngm, bnbt, bnmn, bnvr, out);
}
